// AttentionBlock_39118562132596
// MI455X (gfx1250) — matrix operand order repaired and checked by reading, not compiled, not run on hardware
//
#include <hip/hip_runtime.h>
#include <hip/hip_bf16.h>

// ---------------------------------------------------------------------------
// MI455X (gfx1250) attention block.
// Compute-bound (~100 GFLOP, intermediates fit in 192MB L2) -> everything
// routed through v_wmma_f32_16x16x32_bf16 (wave32, 16x16x32 fragments).
// Per-wave register tile = 32x64 (8 accumulators): 8 WMMA per 12 b128 loads.
// Softmax row staging uses GLOBAL_LOAD_ASYNC_TO_LDS_B128 via inline asm.
// ---------------------------------------------------------------------------

typedef __attribute__((ext_vector_type(16))) __bf16 v16bf;
typedef __attribute__((ext_vector_type(8)))  __bf16 v8bf;
typedef __attribute__((ext_vector_type(8)))  float  v8f;
typedef unsigned short u16;

#define HW   4096              // 64*64 tokens per batch
#define C    512
#define B    2
#define ROWS (B*HW)            // 8192 fused rows for per-pixel linears

__device__ __forceinline__ unsigned f2u(float f){ unsigned u; __builtin_memcpy(&u,&f,4); return u; }
__device__ __forceinline__ float    u2f(unsigned u){ float f; __builtin_memcpy(&f,&u,4); return f; }

// fp32 -> bf16 (round to nearest even), stored as raw u16
__device__ __forceinline__ u16 f2bf(float f){
  unsigned u = f2u(f);
  unsigned r = u + 0x7FFFu + ((u>>16)&1u);
  return (u16)(r>>16);
}
__device__ __forceinline__ float bf2f(u16 h){ return u2f(((unsigned)h)<<16); }

__device__ __forceinline__ v8f wmma_bf16(v16bf a, v16bf b, v8f c){
  // D = A(16x32) * B(32x16) + C, fp32 accumulate
  return __builtin_amdgcn_wmma_f32_16x16x32_bf16(false, a, false, b, (short)0, c, false, false);
}

// A-fragment (16x32 bf16) from a row-major [M x ld] bf16 matrix.
// Contiguous-K layout used here: lanes 0-15 row=lane, elems 0-15 -> K=k0+0..15;
//             lanes 16-31 row=lane-16, elems 0-15 -> K=k0+16..31.
__device__ __forceinline__ v16bf load_a_frag(const u16* A, int ld, int m0, int k0, int lane){
  int row  = m0 + (lane & 15);
  int koff = (lane < 16) ? 0 : 16;
  const u16* base = A + (size_t)row*ld + k0 + koff;
  v8bf lo = *(const v8bf*)(const void*)(base);
  v8bf hi = *(const v8bf*)(const void*)(base + 8);
  return __builtin_shufflevector(lo, hi, 0,1,2,3,4,5,6,7,8,9,10,11,12,13,14,15);
}

// B-fragment (32x16 bf16) where B[k][n] = Src[n][k], Src row-major [N x ldk].
// Contiguous-K layout used here (same as A): lane -> column n0+(lane&15); elems = 16 contiguous K (lanes>=16 offset +16).
__device__ __forceinline__ v16bf load_b_frag_rowK(const u16* Src, int ldk, int n0, int k0, int lane){
  int n    = n0 + (lane & 15);
  int koff = (lane < 16) ? 0 : 16;
  return *(const v16bf*)(const void*)(Src + (size_t)n*ldk + k0 + koff);
}

// B-fragment from pre-swizzled weights: block (bk,bn), 32 lanes x 16 elems contiguous.
__device__ __forceinline__ v16bf load_b_frag_sw(const u16* Wsw, int bk, int bn, int lane){
  return *(const v16bf*)(const void*)(Wsw + (((size_t)(bk*32 + bn)*32) + lane)*16);
}

// ---------------------------------------------------------------------------
// K0: weight prep: fp32 [C x C] row-major -> bf16 swizzled B-fragment layout
// ---------------------------------------------------------------------------
__global__ __launch_bounds__(256) void wprep_kernel(const float* __restrict__ W, u16* __restrict__ Wsw){
  int idx = blockIdx.x*256 + threadIdx.x;       // 512*512 elements
  if (idx >= C*C) return;
  int e    = idx & 15;
  int lane = (idx >> 4) & 31;
  int blk  = idx >> 9;                          // bk*32 + bn
  int bn = blk & 31, bk = blk >> 5;
  int K = bk*32 + ((lane < 16) ? e : 16 + e);
  int N = bn*16 + (lane & 15);
  Wsw[idx] = f2bf(W[K*C + N]);
}

// ---------------------------------------------------------------------------
// K1: GroupNorm stats: one block per (batch, group), reduce 4096*16 elements
// ---------------------------------------------------------------------------
__global__ __launch_bounds__(256) void gn_stats_kernel(const float* __restrict__ x, float* __restrict__ stats){
  int bg = blockIdx.x;                 // b*32+g
  int b = bg >> 5, g = bg & 31;
  const float* xb = x + (size_t)b*HW*C + g*16;
  float s = 0.f, ss = 0.f;
  for (int i = threadIdx.x; i < HW*16; i += 256){
    int p = i >> 4, cc = i & 15;
    float v = xb[(size_t)p*C + cc];
    s += v; ss += v*v;
  }
  __shared__ float rs[256], rq[256];
  rs[threadIdx.x] = s; rq[threadIdx.x] = ss; __syncthreads();
  for (int st = 128; st > 0; st >>= 1){
    if (threadIdx.x < st){ rs[threadIdx.x] += rs[threadIdx.x+st]; rq[threadIdx.x] += rq[threadIdx.x+st]; }
    __syncthreads();
  }
  if (threadIdx.x == 0){
    const float inv_n = 1.0f/(HW*16);
    float mean = rs[0]*inv_n;
    float var  = rq[0]*inv_n - mean*mean;
    stats[bg*2+0] = mean;
    stats[bg*2+1] = rsqrtf(var + 1e-5f);
  }
}

// ---------------------------------------------------------------------------
// K2: normalize + affine, fp32 -> bf16
// ---------------------------------------------------------------------------
__global__ __launch_bounds__(256) void gn_apply_kernel(const float* __restrict__ x,
                                                       const float* __restrict__ gamma,
                                                       const float* __restrict__ beta,
                                                       const float* __restrict__ stats,
                                                       u16* __restrict__ hn){
  int idx = blockIdx.x*256 + threadIdx.x;       // ROWS*C = 2^22
  if (idx >= ROWS*C) return;
  int c = idx & (C-1);
  int b = idx >> 21;                            // HW*C = 2^21
  int g = c >> 4;
  float mean = stats[(b*32+g)*2+0];
  float rstd = stats[(b*32+g)*2+1];
  hn[idx] = f2bf((x[idx]-mean)*rstd*gamma[c] + beta[c]);
}

// ---------------------------------------------------------------------------
// K3: projection GEMM  Out[ROWS x C] = A[ROWS x C] * W + bias
// 4 waves/block; wave computes a 32x64 tile: 2 A-frags x 4 B-frags = 8 WMMA
// per k-step. storeT=0: row-major bf16; storeT=1: transposed vt[b][n][i].
// ---------------------------------------------------------------------------
__global__ __launch_bounds__(128) void gemm_proj_kernel(const u16* __restrict__ A,
                                                        const u16* __restrict__ Wsw,
                                                        const float* __restrict__ bias,
                                                        u16* __restrict__ Out, int storeT){
  int lane = threadIdx.x & 31, wave = threadIdx.x >> 5;
  int m0 = blockIdx.x*128 + wave*32;
  int n0 = blockIdx.y*64;
  v8f acc[2][4] = {};
  for (int k0 = 0; k0 < C; k0 += 32){
    v16bf a0 = load_a_frag(A, C, m0,      k0, lane);
    v16bf a1 = load_a_frag(A, C, m0 + 16, k0, lane);
    int bk = k0 >> 5;
    #pragma unroll
    for (int t = 0; t < 4; ++t){
      v16bf bf = load_b_frag_sw(Wsw, bk, (n0 >> 4) + t, lane);
      acc[0][t] = wmma_bf16(a0, bf, acc[0][t]);
      acc[1][t] = wmma_bf16(a1, bf, acc[1][t]);
    }
  }
  int nc = lane & 15;
  #pragma unroll
  for (int h = 0; h < 2; ++h){
    int mrow = m0 + h*16 + ((lane < 16) ? 0 : 8);
    #pragma unroll
    for (int t = 0; t < 4; ++t){
      int n = n0 + t*16 + nc;
      float bv = bias[n];
      #pragma unroll
      for (int v = 0; v < 8; ++v){
        int m = mrow + v;
        u16 val = f2bf(acc[h][t][v] + bv);
        if (!storeT) Out[(size_t)m*C + n] = val;
        else { int bb = m >> 12, i = m & (HW-1); Out[((size_t)bb*C + n)*HW + i] = val; }
      }
    }
  }
}

// ---------------------------------------------------------------------------
// K4a: scores  S[b][i][j] = (q_i . k_j) * C^-0.5   (bf16 out)
// ---------------------------------------------------------------------------
__global__ __launch_bounds__(128) void scores_kernel(const u16* __restrict__ Q,
                                                     const u16* __restrict__ K,
                                                     u16* __restrict__ S){
  int b = blockIdx.z;
  const u16* Qb = Q + (size_t)b*HW*C;
  const u16* Kb = K + (size_t)b*HW*C;
  u16* Sb = S + (size_t)b*HW*HW;
  int lane = threadIdx.x & 31, wave = threadIdx.x >> 5;
  int i0 = blockIdx.x*128 + wave*32;
  int j0 = blockIdx.y*64;
  v8f acc[2][4] = {};
  for (int c0 = 0; c0 < C; c0 += 32){
    v16bf a0 = load_a_frag(Qb, C, i0,      c0, lane);
    v16bf a1 = load_a_frag(Qb, C, i0 + 16, c0, lane);
    if (c0 + 128 < C){  // stream-ahead hint -> global_prefetch_b8
      __builtin_prefetch(Qb + (size_t)(i0 + (lane & 15))*C + c0 + 128, 0, 0);
    }
    #pragma unroll
    for (int t = 0; t < 4; ++t){
      v16bf bf = load_b_frag_rowK(Kb, C, j0 + t*16, c0, lane);  // B[c][j] = k[j][c]
      acc[0][t] = wmma_bf16(a0, bf, acc[0][t]);
      acc[1][t] = wmma_bf16(a1, bf, acc[1][t]);
    }
  }
  const float scale = 0.04419417382415922f;   // 512^-0.5
  int jc = lane & 15;
  #pragma unroll
  for (int h = 0; h < 2; ++h){
    int irow = i0 + h*16 + ((lane < 16) ? 0 : 8);
    #pragma unroll
    for (int t = 0; t < 4; ++t)
      #pragma unroll
      for (int v = 0; v < 8; ++v)
        Sb[(size_t)(irow+v)*HW + j0 + t*16 + jc] = f2bf(acc[h][t][v]*scale);
  }
}

// ---------------------------------------------------------------------------
// K4b: row softmax in-place over S (row cached in LDS -> 1 read + 1 write).
// Row staging: gfx1250 GLOBAL_LOAD_ASYNC_TO_LDS_B128 (ASYNCcnt) via inline asm.
//   vdst = LDS byte address VGPR; vaddr = 64-bit global address; saddr = off.
// Generic pointers to __shared__ carry the LDS offset in their low 32 bits.
// ---------------------------------------------------------------------------
__global__ __launch_bounds__(256) void softmax_kernel(u16* __restrict__ S){
  __shared__ __align__(16) u16 rowbf[HW];   // 8KB raw bf16 row
  __shared__ float row[HW];                 // 16KB exp values
  __shared__ float red[256];
  u16* Sr = S + (size_t)blockIdx.x*HW;
  int t = threadIdx.x;

  {
    unsigned lds0 = (unsigned)(size_t)(void*)rowbf;   // low 32 bits = LDS offset
    #pragma unroll
    for (int it = 0; it < 2; ++it){                   // 8192B / (256 thr * 16B) = 2
      int cbase = it*4096 + t*16;
      unsigned long long ga = (unsigned long long)(size_t)((const char*)Sr + cbase);
      unsigned lds_addr = lds0 + (unsigned)cbase;
      asm volatile("global_load_async_to_lds_b128 %0, %1, off"
                   :: "v"(lds_addr), "v"(ga) : "memory");
    }
    asm volatile("s_wait_asynccnt 0x0" ::: "memory");
    __syncthreads();
  }

  float mx = -3.0e38f;
  for (int j = t; j < HW; j += 256) mx = fmaxf(mx, bf2f(rowbf[j]));
  red[t] = mx; __syncthreads();
  for (int s = 128; s > 0; s >>= 1){ if (t < s) red[t] = fmaxf(red[t], red[t+s]); __syncthreads(); }
  mx = red[0]; __syncthreads();

  float sum = 0.f;
  for (int j = t; j < HW; j += 256){ float e = __expf(bf2f(rowbf[j])-mx); row[j] = e; sum += e; }
  red[t] = sum; __syncthreads();
  for (int s = 128; s > 0; s >>= 1){ if (t < s) red[t] += red[t+s]; __syncthreads(); }
  float inv = 1.0f/red[0];
  for (int j = t; j < HW; j += 256) Sr[j] = f2bf(row[j]*inv);
}

// ---------------------------------------------------------------------------
// K4c: PV GEMM  o[b][i][n] = sum_j attn[b][i][j] * v[b][j][n]  (uses vt for B)
// ---------------------------------------------------------------------------
__global__ __launch_bounds__(128) void pv_kernel(const u16* __restrict__ Attn,
                                                 const u16* __restrict__ Vt,
                                                 u16* __restrict__ O){
  int b = blockIdx.z;
  const u16* Ab  = Attn + (size_t)b*HW*HW;
  const u16* Vtb = Vt   + (size_t)b*C*HW;     // vt[n][j]
  int lane = threadIdx.x & 31, wave = threadIdx.x >> 5;
  int i0 = blockIdx.x*128 + wave*32;
  int n0 = blockIdx.y*64;
  v8f acc[2][4] = {};
  for (int j0 = 0; j0 < HW; j0 += 32){
    v16bf a0 = load_a_frag(Ab, HW, i0,      j0, lane);
    v16bf a1 = load_a_frag(Ab, HW, i0 + 16, j0, lane);
    if (j0 + 128 < HW){  // attn rows are 8KB, pure streaming -> prefetch ahead
      __builtin_prefetch(Ab + (size_t)(i0 + (lane & 15))*HW + j0 + 128, 0, 0);
    }
    #pragma unroll
    for (int t = 0; t < 4; ++t){
      v16bf bf = load_b_frag_rowK(Vtb, HW, n0 + t*16, j0, lane);  // B[j][n] = vt[n][j]
      acc[0][t] = wmma_bf16(a0, bf, acc[0][t]);
      acc[1][t] = wmma_bf16(a1, bf, acc[1][t]);
    }
  }
  int nc = lane & 15;
  #pragma unroll
  for (int h = 0; h < 2; ++h){
    int irow = i0 + h*16 + ((lane < 16) ? 0 : 8);
    #pragma unroll
    for (int t = 0; t < 4; ++t)
      #pragma unroll
      for (int v = 0; v < 8; ++v)
        O[((size_t)b*HW + irow + v)*C + n0 + t*16 + nc] = f2bf(acc[h][t][v]);
  }
}

// ---------------------------------------------------------------------------
// K5: final projection + residual: out = x + o @ Wp + bp   (fp32 out)
// ---------------------------------------------------------------------------
__global__ __launch_bounds__(128) void final_proj_kernel(const u16* __restrict__ Oc,
                                                         const u16* __restrict__ Wsw,
                                                         const float* __restrict__ bias,
                                                         const float* __restrict__ x,
                                                         float* __restrict__ out){
  int lane = threadIdx.x & 31, wave = threadIdx.x >> 5;
  int m0 = blockIdx.x*128 + wave*32;
  int n0 = blockIdx.y*64;
  v8f acc[2][4] = {};
  for (int k0 = 0; k0 < C; k0 += 32){
    v16bf a0 = load_a_frag(Oc, C, m0,      k0, lane);
    v16bf a1 = load_a_frag(Oc, C, m0 + 16, k0, lane);
    int bk = k0 >> 5;
    #pragma unroll
    for (int t = 0; t < 4; ++t){
      v16bf bf = load_b_frag_sw(Wsw, bk, (n0 >> 4) + t, lane);
      acc[0][t] = wmma_bf16(a0, bf, acc[0][t]);
      acc[1][t] = wmma_bf16(a1, bf, acc[1][t]);
    }
  }
  int nc = lane & 15;
  #pragma unroll
  for (int h = 0; h < 2; ++h){
    int mrow = m0 + h*16 + ((lane < 16) ? 0 : 8);
    #pragma unroll
    for (int t = 0; t < 4; ++t){
      int n = n0 + t*16 + nc;
      float bv = bias[n];
      #pragma unroll
      for (int v = 0; v < 8; ++v){
        size_t idx = (size_t)(mrow+v)*C + n;
        out[idx] = x[idx] + acc[h][t][v] + bv;
      }
    }
  }
}

// ---------------------------------------------------------------------------
extern "C" void kernel_launch(void* const* d_in, const int* in_sizes, int n_in,
                              void* d_out, int out_size, void* d_ws, size_t ws_size,
                              hipStream_t stream) {
  (void)in_sizes; (void)n_in; (void)out_size; (void)ws_size;

  const float* x     = (const float*)d_in[0];
  const float* gamma = (const float*)d_in[1];
  const float* beta  = (const float*)d_in[2];
  const float* W[4]    = {(const float*)d_in[3], (const float*)d_in[5],
                          (const float*)d_in[7], (const float*)d_in[9]};
  const float* bias[4] = {(const float*)d_in[4], (const float*)d_in[6],
                          (const float*)d_in[8], (const float*)d_in[10]};

  // workspace carve-up (256B aligned)
  size_t off = 0;
  auto carve = [&](size_t bytes)->char*{
    char* p = (char*)d_ws + off;
    off += (bytes + 255) & ~(size_t)255;
    return p;
  };
  float* stats = (float*)carve(64*2*sizeof(float));
  u16*   wsw[4];
  for (int i = 0; i < 4; ++i) wsw[i] = (u16*)carve((size_t)C*C*2);
  u16* hn = (u16*)carve((size_t)ROWS*C*2);
  u16* q  = (u16*)carve((size_t)ROWS*C*2);
  u16* k  = (u16*)carve((size_t)ROWS*C*2);
  u16* vt = (u16*)carve((size_t)B*C*HW*2);
  u16* o  = (u16*)carve((size_t)ROWS*C*2);
  u16* S  = (u16*)carve((size_t)B*HW*HW*2);   // 64 MiB scores/attn

  for (int i = 0; i < 4; ++i)
    wprep_kernel<<<(C*C+255)/256, 256, 0, stream>>>(W[i], wsw[i]);

  gn_stats_kernel<<<64, 256, 0, stream>>>(x, stats);
  gn_apply_kernel<<<(ROWS*C+255)/256, 256, 0, stream>>>(x, gamma, beta, stats, hn);

  dim3 gproj(ROWS/128, C/64);
  gemm_proj_kernel<<<gproj, 128, 0, stream>>>(hn, wsw[0], bias[0], q,  0);
  gemm_proj_kernel<<<gproj, 128, 0, stream>>>(hn, wsw[1], bias[1], k,  0);
  gemm_proj_kernel<<<gproj, 128, 0, stream>>>(hn, wsw[2], bias[2], vt, 1);

  scores_kernel<<<dim3(HW/128, HW/64, B), 128, 0, stream>>>(q, k, S);
  softmax_kernel<<<B*HW, 256, 0, stream>>>(S);
  pv_kernel<<<dim3(HW/128, C/64, B), 128, 0, stream>>>(S, vt, o);

  final_proj_kernel<<<gproj, 128, 0, stream>>>(o, wsw[3], bias[3], x, (float*)d_out);
}
